// UnquantizedMoELayer_18287970746807
// MI455X (gfx1250) — matrix operand order repaired and checked by reading, not compiled, not run on hardware
//
#include <hip/hip_runtime.h>

// ---------------------------------------------------------------------------
// MoE FFN for MI455X (gfx1250): grouped bf16 WMMA GEMMs over routed pairs.
// ---------------------------------------------------------------------------

typedef __attribute__((ext_vector_type(16))) __bf16 bf16x16;
typedef __attribute__((ext_vector_type(8)))  __bf16 bf16x8;
typedef __attribute__((ext_vector_type(8)))  float  f32x8;

constexpr int Tt = 4096;
constexpr int Dd = 1024;
constexpr int Ee = 8;
constexpr int Ff = 2816;
constexpr int Kk = 2;

constexpr int BM  = 64;   // rows (pairs) per tile
constexpr int BK  = 32;   // K-step per WMMA stage
constexpr int LSX = 40;   // LDS row stride in bf16 (80B: 16B-aligned, bank-skewed)
constexpr int NPAIR = Tt * Kk;              // 8192
constexpr int MTMAX = NPAIR / BM;           // 128 worst-case row tiles per expert

// ---- helpers ---------------------------------------------------------------

__device__ __forceinline__ bf16x8 cvt8(const float4& a, const float4& b) {
  bf16x8 r;
  r[0] = (__bf16)a.x; r[1] = (__bf16)a.y; r[2] = (__bf16)a.z; r[3] = (__bf16)a.w;
  r[4] = (__bf16)b.x; r[5] = (__bf16)b.y; r[6] = (__bf16)b.z; r[7] = (__bf16)b.w;
  return r;
}

// A fragment (16x32 bf16): lane L<16 -> M=L, K = {kb..kb+7, kb+16..kb+23}, kb = (L>=16)?8:0
__device__ __forceinline__ bf16x16 ld_fragA(const __bf16* tile, int row0, int lane) {
  const __bf16* rp = tile + (row0 + (lane & 15)) * LSX;
  int kb = (lane >> 4) * 8;
  bf16x8 lo = *(const bf16x8*)(rp + kb);
  bf16x8 hi = *(const bf16x8*)(rp + kb + 16);
  bf16x16 f;
#pragma unroll
  for (int i = 0; i < 8; ++i) { f[i] = lo[i]; f[i + 8] = hi[i]; }
  return f;
}

// B fragment (32x16 bf16): lane L<16 -> N=L, K = {kb..kb+7, kb+16..kb+23}, kb = (L>=16)?8:0
__device__ __forceinline__ bf16x16 ld_fragB(const __bf16* tile, int row0, int lane) {
  const __bf16* rp = tile + (row0 + (lane & 15)) * LSX;
  int kb = (lane >> 4) * 8;
  bf16x8 lo = *(const bf16x8*)(rp + kb);
  bf16x8 hi = *(const bf16x8*)(rp + kb + 16);
  bf16x16 f;
#pragma unroll
  for (int i = 0; i < 8; ++i) { f[i] = lo[i]; f[i + 8] = hi[i]; }
  return f;
}

#define WMMA_BF16(A, B, C) \
  __builtin_amdgcn_wmma_f32_16x16x32_bf16(false, (A), false, (B), (short)0, (C), false, false)

// ---- routing ---------------------------------------------------------------

__global__ void k_route_init(int* counts, int* fill) {
  int i = threadIdx.x;
  if (i < Ee) { counts[i] = 0; fill[i] = 0; }
}

__global__ void k_route_count(const int* __restrict__ ids, int* counts) {
  int i = blockIdx.x * blockDim.x + threadIdx.x;
  if (i < NPAIR) atomicAdd(&counts[ids[i]], 1);
}

__global__ void k_route_scan(const int* __restrict__ counts, int* offs) {
  if (threadIdx.x == 0) {
    int acc = 0;
    for (int e = 0; e < Ee; ++e) { offs[e] = acc; acc += counts[e]; }
    offs[Ee] = acc;
  }
}

__global__ void k_route_place(const int* __restrict__ ids,
                              const float* __restrict__ wts,
                              const int* __restrict__ offs, int* fill,
                              int* ptok, float* pwgt, int* pslot) {
  int i = blockIdx.x * blockDim.x + threadIdx.x;
  if (i >= NPAIR) return;
  int e = ids[i];
  int pos = offs[e] + atomicAdd(&fill[e], 1);
  ptok[pos] = i / Kk;
  pwgt[pos] = wts[i];
  pslot[i]  = pos;
}

// ---- GEMM1: act[pair, f] = silu(x@Wg^T) * (x@Wu^T), bf16 out ---------------
// grid: (Ff/64, MTMAX, Ee), block 256 (8 wave32)

__global__ __launch_bounds__(256) void moe_gemm1(
    const float* __restrict__ x, const float* __restrict__ gup,
    const int* __restrict__ offs, const int* __restrict__ ptok,
    __bf16* __restrict__ actg) {
  const int e = blockIdx.z;
  const int off = offs[e];
  const int ne  = offs[e + 1] - off;
  const int rowbase = blockIdx.y * BM;
  if (rowbase >= ne) return;
  const int fbase = blockIdx.x * 64;

  __shared__ __align__(16) __bf16 smem[3 * BM * LSX];
  __bf16* sX = smem;
  __bf16* sG = smem + BM * LSX;
  __bf16* sU = smem + 2 * BM * LSX;

  const int tid  = threadIdx.x;
  const int lane = tid & 31;
  const int wave = tid >> 5;
  const int mi = wave & 3;   // 4 m-subtiles of 16
  const int ni = wave >> 2;  // 2 n-subtiles of 32

  // staging: each thread loads 8 fp32 of one row (64 rows x 32 cols)
  const int srow = tid >> 2;
  const int scol = (tid & 3) * 8;
  const int gr = rowbase + srow;
  const int tk = ptok[off + (gr < ne ? gr : 0)];
  const float* xrow = x + (size_t)tk * Dd + scol;
  const float* grow = gup + ((size_t)e * 2 * Ff + (fbase + srow)) * (size_t)Dd + scol;
  const float* urow = gup + ((size_t)e * 2 * Ff + (Ff + fbase + srow)) * (size_t)Dd + scol;

  f32x8 cg0{}, cg1{}, cu0{}, cu1{};
  float4 rx0, rx1, rg0, rg1, ru0, ru1;
  rx0 = *(const float4*)(xrow); rx1 = *(const float4*)(xrow + 4);
  rg0 = *(const float4*)(grow); rg1 = *(const float4*)(grow + 4);
  ru0 = *(const float4*)(urow); ru1 = *(const float4*)(urow + 4);

  const int NK = Dd / BK;
  for (int kt = 0; kt < NK; ++kt) {
    bf16x8 vx = cvt8(rx0, rx1);
    bf16x8 vg = cvt8(rg0, rg1);
    bf16x8 vu = cvt8(ru0, ru1);
    if (kt + 1 < NK) {  // pipeline: next-stage global loads in flight during WMMA
      int k0 = (kt + 1) * BK;
      rx0 = *(const float4*)(xrow + k0); rx1 = *(const float4*)(xrow + k0 + 4);
      rg0 = *(const float4*)(grow + k0); rg1 = *(const float4*)(grow + k0 + 4);
      ru0 = *(const float4*)(urow + k0); ru1 = *(const float4*)(urow + k0 + 4);
    }
    __syncthreads();
    *(bf16x8*)(sX + srow * LSX + scol) = vx;
    *(bf16x8*)(sG + srow * LSX + scol) = vg;
    *(bf16x8*)(sU + srow * LSX + scol) = vu;
    __syncthreads();

    bf16x16 a = ld_fragA(sX, mi * 16, lane);
    bf16x16 b0 = ld_fragB(sG, ni * 32, lane);
    cg0 = WMMA_BF16(a, b0, cg0);
    bf16x16 b1 = ld_fragB(sG, ni * 32 + 16, lane);
    cg1 = WMMA_BF16(a, b1, cg1);
    bf16x16 b2 = ld_fragB(sU, ni * 32, lane);
    cu0 = WMMA_BF16(a, b2, cu0);
    bf16x16 b3 = ld_fragB(sU, ni * 32 + 16, lane);
    cu1 = WMMA_BF16(a, b3, cu1);
  }

  // silu(gate)*up -> LDS transpose -> coalesced bf16x8 global stores
  __syncthreads();
  __bf16* sAct = smem;  // 64 x 72 bf16 (reuse staging LDS)
  const int mhi = (lane >> 4) << 3;
  const int nc  = lane & 15;
#pragma unroll
  for (int j = 0; j < 8; ++j) {
    int m = mi * 16 + j + mhi;
    float g0 = cg0[j], g1 = cg1[j];
    float a0 = (g0 / (1.f + __expf(-g0))) * cu0[j];
    float a1 = (g1 / (1.f + __expf(-g1))) * cu1[j];
    sAct[m * 72 + ni * 32 + nc]      = (__bf16)a0;
    sAct[m * 72 + ni * 32 + 16 + nc] = (__bf16)a1;
  }
  __syncthreads();
#pragma unroll
  for (int it = 0; it < 2; ++it) {
    int chunk = tid + it * 256;         // 512 chunks of 8 bf16
    int r = chunk >> 3, c = (chunk & 7) * 8;
    if (rowbase + r < ne)
      *(bf16x8*)(actg + (size_t)(off + rowbase + r) * Ff + fbase + c) =
          *(const bf16x8*)(sAct + r * 72 + c);
  }
}

// ---- GEMM2: outp[pair, d] = wgt * (act @ Wd^T) -----------------------------
// grid: (Dd/64, MTMAX, Ee), block 256

__global__ __launch_bounds__(256) void moe_gemm2(
    const __bf16* __restrict__ actg, const float* __restrict__ down,
    const int* __restrict__ offs, const float* __restrict__ pwgt,
    float* __restrict__ outp) {
  const int e = blockIdx.z;
  const int off = offs[e];
  const int ne  = offs[e + 1] - off;
  const int rowbase = blockIdx.y * BM;
  if (rowbase >= ne) return;
  const int dbase = blockIdx.x * 64;

  __shared__ __align__(16) __bf16 sA[BM * LSX];
  __shared__ __align__(16) __bf16 sB[BM * LSX];

  const int tid  = threadIdx.x;
  const int lane = tid & 31;
  const int wave = tid >> 5;
  const int mi = wave & 3, ni = wave >> 2;

  const int srow = tid >> 2;
  const int scol = (tid & 3) * 8;
  const int gr = rowbase + srow;
  const int arow = off + (gr < ne ? gr : 0);
  const __bf16* arp = actg + (size_t)arow * Ff + scol;
  const float*  brp = down + ((size_t)e * Dd + (dbase + srow)) * (size_t)Ff + scol;

  f32x8 c0{}, c1{};
  bf16x8 ra = *(const bf16x8*)(arp);
  float4 rb0 = *(const float4*)(brp);
  float4 rb1 = *(const float4*)(brp + 4);

  const int NK = Ff / BK;  // 88
  for (int kt = 0; kt < NK; ++kt) {
    bf16x8 va = ra;
    bf16x8 vb = cvt8(rb0, rb1);
    if (kt + 1 < NK) {
      int k0 = (kt + 1) * BK;
      ra  = *(const bf16x8*)(arp + k0);
      rb0 = *(const float4*)(brp + k0);
      rb1 = *(const float4*)(brp + k0 + 4);
    }
    __syncthreads();
    *(bf16x8*)(sA + srow * LSX + scol) = va;
    *(bf16x8*)(sB + srow * LSX + scol) = vb;
    __syncthreads();

    bf16x16 a = ld_fragA(sA, mi * 16, lane);
    bf16x16 b0 = ld_fragB(sB, ni * 32, lane);
    c0 = WMMA_BF16(a, b0, c0);
    bf16x16 b1 = ld_fragB(sB, ni * 32 + 16, lane);
    c1 = WMMA_BF16(a, b1, c1);
  }

  const int mhi = (lane >> 4) << 3;
  const int nc  = lane & 15;
  float w[8];
#pragma unroll
  for (int j = 0; j < 8; ++j) {
    int r = rowbase + mi * 16 + j + mhi;
    w[j] = (r < ne) ? pwgt[off + r] : 0.f;
  }
#pragma unroll
  for (int j = 0; j < 8; ++j) {
    int r = rowbase + mi * 16 + j + mhi;
    if (r < ne) {
      size_t base = (size_t)(off + r) * Dd + dbase + ni * 32;
      outp[base + nc]      = c0[j] * w[j];
      outp[base + 16 + nc] = c1[j] * w[j];
    }
  }
}

// ---- combine: out[t,d] = sum_k outp[slot(t,k), d] (non-atomic, deterministic)

__global__ void k_combine(const float* __restrict__ outp,
                          const int* __restrict__ pslot,
                          float* __restrict__ out) {
  int idx = blockIdx.x * blockDim.x + threadIdx.x;
  if (idx >= Tt * Dd) return;
  int t = idx >> 10;          // Dd = 1024
  int d = idx & (Dd - 1);
  int s0 = pslot[t * Kk];
  int s1 = pslot[t * Kk + 1];
  out[idx] = outp[(size_t)s0 * Dd + d] + outp[(size_t)s1 * Dd + d];
}

// ---------------------------------------------------------------------------

extern "C" void kernel_launch(void* const* d_in, const int* in_sizes, int n_in,
                              void* d_out, int out_size, void* d_ws, size_t ws_size,
                              hipStream_t stream) {
  (void)in_sizes; (void)n_in; (void)out_size; (void)ws_size;
  const float* x    = (const float*)d_in[0];
  const float* gup  = (const float*)d_in[1];
  const float* down = (const float*)d_in[2];
  const float* wts  = (const float*)d_in[3];
  const int*   ids  = (const int*)d_in[4];
  float* out = (float*)d_out;

  char* ws = (char*)d_ws;
  int*    counts = (int*)(ws + 0);
  int*    fill   = (int*)(ws + 256);
  int*    offs   = (int*)(ws + 512);
  int*    ptok   = (int*)(ws + 1024);
  float*  pwgt   = (float*)(ws + 1024 + 32768);
  int*    pslot  = (int*)(ws + 1024 + 2 * 32768);
  __bf16* actg   = (__bf16*)(ws + 99328);                      // 8192 x 2816 bf16
  float*  outp   = (float*)(ws + 99328 + (size_t)NPAIR * Ff * 2);  // 8192 x 1024 f32

  k_route_init<<<1, 32, 0, stream>>>(counts, fill);
  k_route_count<<<(NPAIR + 255) / 256, 256, 0, stream>>>(ids, counts);
  k_route_scan<<<1, 32, 0, stream>>>(counts, offs);
  k_route_place<<<(NPAIR + 255) / 256, 256, 0, stream>>>(ids, wts, offs, fill,
                                                         ptok, pwgt, pslot);
  moe_gemm1<<<dim3(Ff / 64, MTMAX, Ee), 256, 0, stream>>>(x, gup, offs, ptok, actg);
  moe_gemm2<<<dim3(Dd / 64, MTMAX, Ee), 256, 0, stream>>>(actg, down, offs, pwgt, outp);
  k_combine<<<(Tt * Dd + 255) / 256, 256, 0, stream>>>(outp, pslot, out);
}
